// MultiHeadAttention_13091060318508
// MI455X (gfx1250) — compile-verified, matrix operand order checked by reading, not run on hardware
//
#include <hip/hip_runtime.h>
#include <hip/hip_bf16.h>

typedef __bf16 v16bf __attribute__((ext_vector_type(16)));
typedef float  v8f   __attribute__((ext_vector_type(8)));
typedef int    v4i   __attribute__((ext_vector_type(4)));

#define AS1 __attribute__((address_space(1)))
#define AS3 __attribute__((address_space(3)))

// gfx1250 async global->LDS path (ASYNCcnt-tracked), with safe fallback.
#if __has_builtin(__builtin_amdgcn_global_load_async_to_lds_b128) && \
    __has_builtin(__builtin_amdgcn_s_wait_asynccnt)
#define ASYNC_LDS 1
#endif

// ---------------------------------------------------------------- helpers

__device__ __forceinline__ __bf16 f2bf(float f) {
    unsigned u = __float_as_uint(f);
    unsigned r = (u + 0x7fffu + ((u >> 16) & 1u)) >> 16;
    unsigned short s = (unsigned short)r;
    return __builtin_bit_cast(__bf16, s);
}

// Build a WMMA A/B bf16 fragment from a row-major [16][rowStride] bf16 tile.
// ISA layout (CDNA5 ISA):
//   lanes 0-15 : row = lane,    K = {0..7, 16..23}
//   lanes 16-31: row = lane-16, K = {8..15, 24..31}
__device__ __forceinline__ v16bf frag_from(const __bf16* base, int rowStride, int lane) {
    int m  = lane & 15;
    int kb = (lane >> 4) << 3;           // 0 or 8
    const __bf16* p = base + m * rowStride + kb;
    union { struct { v4i a, b; } s; v16bf f; } u;
    u.s.a = *(const v4i*)(p);
    u.s.b = *(const v4i*)(p + 16);
    return u.f;
}

// Stage 8 contiguous elements into LDS as bf16.
__device__ __forceinline__ void stage8(__bf16* dst, const float* src) {
    const float4* s = (const float4*)src;
    float4 a = s[0], b = s[1];
    union { __bf16 h[8]; v4i v; } u;
    u.h[0] = f2bf(a.x); u.h[1] = f2bf(a.y); u.h[2] = f2bf(a.z); u.h[3] = f2bf(a.w);
    u.h[4] = f2bf(b.x); u.h[5] = f2bf(b.y); u.h[6] = f2bf(b.z); u.h[7] = f2bf(b.w);
    *(v4i*)dst = u.v;
}
__device__ __forceinline__ void stage8(__bf16* dst, const __bf16* src) {
#ifdef ASYNC_LDS
    // 16B global -> LDS, no VGPR roundtrip; tracked by ASYNCcnt.
    // Param 1: AS1 pointer to the 16B element; param 2: AS3 LDS destination
    // (low 32 bits of a generic LDS address are the LDS offset, ISA 10.2).
    __builtin_amdgcn_global_load_async_to_lds_b128(
        (AS1 v4i*)(uintptr_t)src,
        (AS3 v4i*)(uint32_t)(uintptr_t)dst, 0, 0);
#else
    *(v4i*)dst = *(const v4i*)src;
#endif
}

__device__ __forceinline__ void wait_stage() {
#ifdef ASYNC_LDS
    __builtin_amdgcn_s_wait_asynccnt(0);
#endif
    __syncthreads();
}

// ---------------------------------------------------------------- GEMM
// C[m,n] = sum_k A[m,k] * B[n,k]  (+ bias[n])   -- "x @ W.T" layout.
// Block: 256 threads = 8 waves; 128x64 output tile; K staged 32 at a time.
// Each wave owns a 32x32 sub-tile -> 4 WMMA accumulators, 4 WMMAs / K-step.
// MODE 0: store bf16 [B,H,L,DH]      (Q,K projections)
// MODE 1: store bf16 [B,H,DH,L]      (V projection, transposed)
// MODE 2: store f32  [M,N] row-major (O projection tmp)
// MODE 3: store bf16 [B,L,D], batched over bh = blockIdx.z (ctx = attn @ V)
template <int MODE>
__device__ __forceinline__ void store_one(void* Out, int m, int n, float v, int N,
                                          int batch, int Lsz, int Hsz, int DHsz) {
    const int Dsz = Hsz * DHsz;
    if constexpr (MODE == 0) {            // bf16 [B,H,L,DH]
        __bf16* o = (__bf16*)Out;
        int bb = m / Lsz, l = m % Lsz, h = n / DHsz, dh = n % DHsz;
        o[(((size_t)bb * Hsz + h) * Lsz + l) * DHsz + dh] = f2bf(v);
    } else if constexpr (MODE == 1) {     // bf16 [B,H,DH,L]
        __bf16* o = (__bf16*)Out;
        int bb = m / Lsz, l = m % Lsz, h = n / DHsz, dh = n % DHsz;
        o[(((size_t)bb * Hsz + h) * DHsz + dh) * Lsz + l] = f2bf(v);
    } else if constexpr (MODE == 2) {     // f32 [M,N]
        ((float*)Out)[(size_t)m * N + n] = v;
    } else {                              // MODE 3: bf16 [B,L,D], batch = b*H+h
        __bf16* o = (__bf16*)Out;
        int bb = batch / Hsz, h = batch % Hsz;
        o[((size_t)bb * Lsz + m) * Dsz + h * DHsz + n] = f2bf(v);
    }
}

template <typename AT, typename BT, int MODE>
__global__ __launch_bounds__(256)
void gemm_tn(const AT* __restrict__ Ag, const BT* __restrict__ Bg,
             const float* __restrict__ bias, void* __restrict__ Out,
             int M, int N, int K, int lda, int ldb,
             long long batchStrideA, long long batchStrideB,
             int Bsz, int Lsz, int Hsz, int DHsz)
{
    __shared__ __bf16 As[128 * 32];      // 8 KB
    __shared__ __bf16 Bs[64 * 32];       // 4 KB

    const int tid  = threadIdx.x;
    const int lane = tid & 31;
    const int wave = tid >> 5;
    const int wr = wave >> 1;            // 0..3 -> 32-row band
    const int wc = wave & 1;             // 0..1 -> 32-col band
    const int batch = blockIdx.z;

    const AT* Ab = Ag + (size_t)batch * batchStrideA;
    const BT* Bb = Bg + (size_t)batch * batchStrideB;
    const int m0 = blockIdx.y * 128;
    const int n0 = blockIdx.x * 64;

    v8f c00 = {}, c01 = {}, c10 = {}, c11 = {};

    // A staging: 128x32 = 4096 elems -> 16/thread (two stage8 of 8).
    const int ea   = tid * 16;
    const int arow = ea >> 5;            // 0..127
    const int acol = ea & 31;            // 0 or 16
    // B staging: 64x32 = 2048 elems -> 8/thread.
    const int eb   = tid * 8;
    const int brow = eb >> 5;            // 0..63
    const int bcol = eb & 31;            // 0,8,16,24

    for (int k0 = 0; k0 < K; k0 += 32) {
        const AT* aSrc = Ab + (size_t)(m0 + arow) * lda + k0 + acol;
        const BT* bSrc = Bb + (size_t)(n0 + brow) * ldb + k0 + bcol;
        stage8(As + arow * 32 + acol, aSrc);
        stage8(As + arow * 32 + acol + 8, aSrc + 8);
        stage8(Bs + brow * 32 + bcol, bSrc);
        if (k0 + 32 < K) {               // pull next K-slice toward the caches
            __builtin_prefetch(aSrc + 32, 0, 3);
            __builtin_prefetch(bSrc + 32, 0, 3);
        }
        wait_stage();

        v16bf fa0 = frag_from(As + (wr * 32) * 32, 32, lane);
        v16bf fa1 = frag_from(As + (wr * 32 + 16) * 32, 32, lane);
        v16bf fb0 = frag_from(Bs + (wc * 32) * 32, 32, lane);
        v16bf fb1 = frag_from(Bs + (wc * 32 + 16) * 32, 32, lane);
        c00 = __builtin_amdgcn_wmma_f32_16x16x32_bf16(false, fa0, false, fb0,
                                                      (short)0, c00, false, false);
        c01 = __builtin_amdgcn_wmma_f32_16x16x32_bf16(false, fa0, false, fb1,
                                                      (short)0, c01, false, false);
        c10 = __builtin_amdgcn_wmma_f32_16x16x32_bf16(false, fa1, false, fb0,
                                                      (short)0, c10, false, false);
        c11 = __builtin_amdgcn_wmma_f32_16x16x32_bf16(false, fa1, false, fb1,
                                                      (short)0, c11, false, false);
        __syncthreads();
    }

    // C layout: VGPR r -> M = r (lanes 0-15) / r+8 (lanes 16-31), N = lane&15
    const int nc0 = n0 + wc * 32 + (lane & 15);
    const int nc1 = nc0 + 16;
    const float b0 = bias ? bias[nc0] : 0.0f;
    const float b1 = bias ? bias[nc1] : 0.0f;
    const int mA = m0 + wr * 32 + ((lane >> 4) ? 8 : 0);
    const int mB = mA + 16;

    for (int r = 0; r < 8; ++r) {
        store_one<MODE>(Out, mA + r, nc0, c00[r] + b0, N, batch, Lsz, Hsz, DHsz);
        store_one<MODE>(Out, mA + r, nc1, c01[r] + b1, N, batch, Lsz, Hsz, DHsz);
        store_one<MODE>(Out, mB + r, nc0, c10[r] + b0, N, batch, Lsz, Hsz, DHsz);
        store_one<MODE>(Out, mB + r, nc1, c11[r] + b1, N, batch, Lsz, Hsz, DHsz);
    }
}

// ---------------------------------------------------------------- scores
// S[bh, qi*16+.., kj*16+..] = scale * Q[bh] . K[bh]^T ; one 16x16 tile / wave.
__global__ __launch_bounds__(256)
void scores_kernel(const __bf16* __restrict__ Q, const __bf16* __restrict__ Kt,
                   float* __restrict__ S, float scale)
{
    const int lane = threadIdx.x & 31;
    const int wave = threadIdx.x >> 5;
    int tile = blockIdx.x * 8 + wave;          // 128 * 64 * 64 tiles
    const int kj = tile & 63; tile >>= 6;
    const int qi = tile & 63; tile >>= 6;
    const int bh = tile;                       // 0..127

    const __bf16* Qb = Q  + ((size_t)bh * 1024 + qi * 16) * 64;
    const __bf16* Kb = Kt + ((size_t)bh * 1024 + kj * 16) * 64;

    v8f acc = {};
    for (int k0 = 0; k0 < 64; k0 += 32) {
        v16bf fa = frag_from(Qb + k0, 64, lane);   // A: rows = queries
        v16bf fb = frag_from(Kb + k0, 64, lane);   // B: lane n = key n along k
        acc = __builtin_amdgcn_wmma_f32_16x16x32_bf16(false, fa, false, fb,
                                                      (short)0, acc, false, false);
    }

    float* Sb = S + (((size_t)bh * 1024 + qi * 16) * 1024) + kj * 16;
    const int rbase = (lane >> 4) ? 8 : 0;
    const int cidx  = lane & 15;
    for (int r = 0; r < 8; ++r)
        Sb[(size_t)(rbase + r) * 1024 + cidx] = acc[r] * scale;
}

// ---------------------------------------------------------------- softmax
// One wave per row of 1024 floats (in place).
__global__ __launch_bounds__(256)
void softmax_kernel(float* __restrict__ S)
{
    const int lane = threadIdx.x & 31;
    const int wave = threadIdx.x >> 5;
    const size_t row = (size_t)blockIdx.x * 8 + wave;
    float* p = S + row * 1024;

    float4 v[8];
    float mx = -3.4e38f;
    for (int i = 0; i < 8; ++i) {
        v[i] = *(const float4*)(p + lane * 4 + i * 128);
        mx = fmaxf(mx, fmaxf(fmaxf(v[i].x, v[i].y), fmaxf(v[i].z, v[i].w)));
    }
    for (int off = 16; off; off >>= 1) mx = fmaxf(mx, __shfl_xor(mx, off, 32));

    float sum = 0.0f;
    for (int i = 0; i < 8; ++i) {
        v[i].x = __expf(v[i].x - mx); v[i].y = __expf(v[i].y - mx);
        v[i].z = __expf(v[i].z - mx); v[i].w = __expf(v[i].w - mx);
        sum += v[i].x + v[i].y + v[i].z + v[i].w;
    }
    for (int off = 16; off; off >>= 1) sum += __shfl_xor(sum, off, 32);

    const float inv = 1.0f / sum;
    for (int i = 0; i < 8; ++i) {
        v[i].x *= inv; v[i].y *= inv; v[i].z *= inv; v[i].w *= inv;
        *(float4*)(p + lane * 4 + i * 128) = v[i];
    }
}

// ---------------------------------------------------------------- layernorm
// out = LN(otmp + residual) * gamma + beta ; one wave per row of 1024.
__global__ __launch_bounds__(256)
void layernorm_kernel(const float* __restrict__ Otmp, const float* __restrict__ resid,
                      const float* __restrict__ gamma, const float* __restrict__ beta,
                      float* __restrict__ out)
{
    const int lane = threadIdx.x & 31;
    const int wave = threadIdx.x >> 5;
    const size_t row = (size_t)blockIdx.x * 8 + wave;
    const float* po = Otmp  + row * 1024;
    const float* pr = resid + row * 1024;

    float4 v[8];
    float sum = 0.0f, sumsq = 0.0f;
    for (int i = 0; i < 8; ++i) {
        float4 a = *(const float4*)(po + lane * 4 + i * 128);
        float4 b = *(const float4*)(pr + lane * 4 + i * 128);
        v[i].x = a.x + b.x; v[i].y = a.y + b.y; v[i].z = a.z + b.z; v[i].w = a.w + b.w;
        sum   += v[i].x + v[i].y + v[i].z + v[i].w;
        sumsq += v[i].x * v[i].x + v[i].y * v[i].y + v[i].z * v[i].z + v[i].w * v[i].w;
    }
    for (int off = 16; off; off >>= 1) {
        sum   += __shfl_xor(sum, off, 32);
        sumsq += __shfl_xor(sumsq, off, 32);
    }
    const float mu  = sum * (1.0f / 1024.0f);
    const float var = sumsq * (1.0f / 1024.0f) - mu * mu;
    const float rs  = rsqrtf(var + 1e-5f);

    float* pw = out + row * 1024;
    for (int i = 0; i < 8; ++i) {
        const int c = lane * 4 + i * 128;
        float4 g = *(const float4*)(gamma + c);
        float4 t = *(const float4*)(beta + c);
        float4 o;
        o.x = (v[i].x - mu) * rs * g.x + t.x;
        o.y = (v[i].y - mu) * rs * g.y + t.y;
        o.z = (v[i].z - mu) * rs * g.z + t.z;
        o.w = (v[i].w - mu) * rs * g.w + t.w;
        *(float4*)(pw + c) = o;
    }
}

// ---------------------------------------------------------------- launch

extern "C" void kernel_launch(void* const* d_in, const int* in_sizes, int n_in,
                              void* d_out, int out_size, void* d_ws, size_t ws_size,
                              hipStream_t stream) {
    const float* key   = (const float*)d_in[0];
    const float* value = (const float*)d_in[1];
    const float* query = (const float*)d_in[2];
    const float* Wq = (const float*)d_in[3];
    const float* bq = (const float*)d_in[4];
    const float* Wk = (const float*)d_in[5];
    const float* bk = (const float*)d_in[6];
    const float* Wv = (const float*)d_in[7];
    const float* bv = (const float*)d_in[8];
    const float* Wo = (const float*)d_in[9];
    const float* bo = (const float*)d_in[10];
    const float* gamma = (const float*)d_in[11];
    const float* beta  = (const float*)d_in[12];

    constexpr int Bsz = 8, Lsz = 1024, Dsz = 1024, Hsz = 16, DHsz = 64;
    constexpr int Mrows = Bsz * Lsz;                       // 8192
    const size_t BLD = (size_t)Bsz * Lsz * Dsz;            // 8388608

    float* out  = (float*)d_out;
    float* attn = out + BLD;                               // [B,H,L,L] region of d_out

    // workspace layout (64 MB): q(16M) k(16M) vt(16M) ctx(16M); otmp(32M f32)
    // aliases the q+k region, which is dead by the time the O-projection runs.
    char* ws = (char*)d_ws;
    __bf16* q_bf   = (__bf16*)(ws);
    __bf16* k_bf   = (__bf16*)(ws + (16u << 20));
    __bf16* vt_bf  = (__bf16*)(ws + (32u << 20));
    __bf16* ctx_bf = (__bf16*)(ws + (48u << 20));
    float*  otmp   = (float*)(ws);

    dim3 blk(256);

    // Q,K projections -> bf16 [B,H,L,DH] ; V -> bf16 [B,H,DH,L]
    gemm_tn<float, float, 0><<<dim3(16, 64, 1), blk, 0, stream>>>(
        query, Wq, bq, q_bf, Mrows, Dsz, Dsz, Dsz, Dsz, 0, 0, Bsz, Lsz, Hsz, DHsz);
    gemm_tn<float, float, 0><<<dim3(16, 64, 1), blk, 0, stream>>>(
        key, Wk, bk, k_bf, Mrows, Dsz, Dsz, Dsz, Dsz, 0, 0, Bsz, Lsz, Hsz, DHsz);
    gemm_tn<float, float, 1><<<dim3(16, 64, 1), blk, 0, stream>>>(
        value, Wv, bv, vt_bf, Mrows, Dsz, Dsz, Dsz, Dsz, 0, 0, Bsz, Lsz, Hsz, DHsz);

    // scores -> attn region of d_out (pre-softmax), then softmax in place
    scores_kernel<<<65536, blk, 0, stream>>>(q_bf, k_bf, attn, 0.125f);
    softmax_kernel<<<16384, blk, 0, stream>>>(attn);

    // ctx = attn @ V : batched over bh, A f32 (bf16-converted on stage), B bf16 (async-LDS)
    gemm_tn<float, __bf16, 3><<<dim3(1, 8, 128), blk, 0, stream>>>(
        attn, vt_bf, nullptr, ctx_bf, Lsz, DHsz, Lsz, Lsz, Lsz,
        (long long)Lsz * Lsz, (long long)DHsz * Lsz, Bsz, Lsz, Hsz, DHsz);

    // O projection (A = ctx bf16 via async-LDS) -> f32 tmp, then residual + LN -> out
    gemm_tn<__bf16, float, 2><<<dim3(16, 64, 1), blk, 0, stream>>>(
        ctx_bf, Wo, bo, otmp, Mrows, Dsz, Dsz, Dsz, Dsz, 0, 0, Bsz, Lsz, Hsz, DHsz);
    layernorm_kernel<<<1024, blk, 0, stream>>>(otmp, query, gamma, beta, out);
}
